// STAPBlock_25555055411398
// MI455X (gfx1250) — compile-verified, matrix operand order checked by reading, not run on hardware
//
#include <hip/hip_runtime.h>
#include <hip/hip_bf16.h>
#include <math.h>

typedef __attribute__((ext_vector_type(16))) _Float16 v16h;
typedef __attribute__((ext_vector_type(8)))  float    v8f;

#define BB    64
#define CC    256
#define TT    128
#define JJ    17
#define NHH   8
#define HDD   32
#define HIDN  1024
#define HCC   43
#define WCC   6
#define PP    258      // HCC*WCC tokens
#define PPAD  288      // padded key length (9 * 32)
#define M1    (BB*PP)       // 16512 rows of qkv GEMM
#define K1    (CC*9)        // 2304
#define N1    (3*CC)        // 768
#define M2    (BB*TT*JJ)    // 139264 rows of 1x1 GEMMs
#define MCH   17408         // M2 / 8 chunks
#define NCHK  8

// ---------------------------------------------------------------- helpers ---
__device__ __forceinline__ float bn_ap(float x, const float* g, const float* b,
                                       const float* m, const float* v, int c) {
  float inv = g[c] * rsqrtf(v[c] + 1e-5f);
  return x * inv + (b[c] - m[c] * inv);
}
__device__ __forceinline__ float gelu_ex(float x) {
  return 0.5f * x * (1.0f + erff(x * 0.70710678118654752f));
}

union FragAB { v16h v; unsigned u[8]; };

// Load a 16x32 f16 fragment (A layout) / 32x16 (B layout, row=N, K contiguous)
// from row-major f16 memory: lane's row = `row`, K(lane,e) per CDNA5 ISA:
// K = (e&7) + 16*(e>=8) + 8*(lane>=16); consecutive (2e,2e+1) pairs -> b32 loads
// which the backend merges into ds_load_b128 / global_load_b128.
__device__ __forceinline__ FragAB frag_ld(const _Float16* base, int row,
                                          int stride, int lane) {
  FragAB f;
  const int hi = (lane >> 4) & 1;
  const _Float16* r = base + (long)row * stride;
#pragma unroll
  for (int p = 0; p < 8; ++p) {
    const int k0 = ((p < 4) ? (2 * p) : (2 * p + 8)) + hi * 8;
    f.u[p] = *(const unsigned*)(r + k0);
  }
  return f;
}

// --------------------------------------------------- shared GEMM main loop ---
// Block tile 64(M) x 128(N), 8 waves in 2x4 grid, each wave owns a 32x32 tile
// (2x2 16x16 f32 accumulators -> 4 WMMAs per 4 fragment loads per K-step).
// Ping-pong LDS double buffer: one barrier per K-step; global->reg prefetch
// overlaps the WMMA window. A:[M][K], Bw:[N][K] row-major f16 (A * Bw^T).
__device__ __forceinline__ void gemm_block(const _Float16* __restrict__ A,
                                           const _Float16* __restrict__ Bw,
                                           int K, v8f (&cacc)[2][2],
                                           int& m0, int& n0) {
  constexpr int AS = 40;  // padded LDS row stride (halves); 80B keeps 16B align
  __shared__ __align__(16) _Float16 sA[2 * 64 * AS];
  __shared__ __align__(16) _Float16 sB[2 * 128 * AS];
  const int tid  = threadIdx.x;
  const int w    = tid >> 5, lane = tid & 31;
  const int wm   = w & 1,    wn   = w >> 1;           // 2 x 4 wave grid
  const long row0 = (long)blockIdx.x * 64;
  const long col0 = (long)blockIdx.y * 128;
  m0 = (int)row0 + wm * 32;
  n0 = (int)col0 + wn * 32;
  const int ar = tid >> 2, ak = (tid & 3) * 8;        // A: 64 rows x 32 halves
  const int br = tid >> 1, bk = (tid & 1) * 16;       // B: 128 rows x 32 halves
#pragma unroll
  for (int mi = 0; mi < 2; ++mi)
#pragma unroll
    for (int ni = 0; ni < 2; ++ni)
#pragma unroll
      for (int i = 0; i < 8; ++i) cacc[mi][ni][i] = 0.0f;

  const _Float16* Ap = A  + (row0 + ar) * (long)K + ak;
  const _Float16* Bp = Bw + (col0 + br) * (long)K + bk;
  float4 ra  = *(const float4*)(Ap);
  float4 rb0 = *(const float4*)(Bp);
  float4 rb1 = *(const float4*)(Bp + 8);
  *(float4*)(sA + ar * AS + ak)     = ra;
  *(float4*)(sB + br * AS + bk)     = rb0;
  *(float4*)(sB + br * AS + bk + 8) = rb1;
  __syncthreads();

  int buf = 0;
  for (int kk = 0; kk < K; kk += 32) {
    const bool nxt = (kk + 32) < K;
    if (nxt) {  // register prefetch of next K-slab overlaps the WMMAs below
      ra  = *(const float4*)(Ap + kk + 32);
      rb0 = *(const float4*)(Bp + kk + 32);
      rb1 = *(const float4*)(Bp + kk + 40);
      if (kk + 64 < K) __builtin_prefetch(Ap + kk + 64, 0, 3);
    }
    const _Float16* cA = sA + buf * (64 * AS);
    const _Float16* cB = sB + buf * (128 * AS);
    FragAB a0 = frag_ld(cA + (wm * 32) * AS,      lane & 15, AS, lane);
    FragAB a1 = frag_ld(cA + (wm * 32 + 16) * AS, lane & 15, AS, lane);
    FragAB b0 = frag_ld(cB + (wn * 32) * AS,      lane & 15, AS, lane);
    FragAB b1 = frag_ld(cB + (wn * 32 + 16) * AS, lane & 15, AS, lane);
    cacc[0][0] = __builtin_amdgcn_wmma_f32_16x16x32_f16(false, a0.v, false, b0.v,
                                                        (short)0, cacc[0][0], false, false);
    cacc[0][1] = __builtin_amdgcn_wmma_f32_16x16x32_f16(false, a0.v, false, b1.v,
                                                        (short)0, cacc[0][1], false, false);
    cacc[1][0] = __builtin_amdgcn_wmma_f32_16x16x32_f16(false, a1.v, false, b0.v,
                                                        (short)0, cacc[1][0], false, false);
    cacc[1][1] = __builtin_amdgcn_wmma_f32_16x16x32_f16(false, a1.v, false, b1.v,
                                                        (short)0, cacc[1][1], false, false);
    if (nxt) {
      _Float16* nA = sA + (buf ^ 1) * (64 * AS);
      _Float16* nB = sB + (buf ^ 1) * (128 * AS);
      *(float4*)(nA + ar * AS + ak)     = ra;
      *(float4*)(nB + br * AS + bk)     = rb0;
      *(float4*)(nB + br * AS + bk + 8) = rb1;
    }
    __syncthreads();
    buf ^= 1;
  }
}

// ----------------------------------------------------------- small kernels ---
__global__ void k_cast_f16(const float* __restrict__ in,
                           _Float16* __restrict__ out, long n) {
  long i = (long)blockIdx.x * blockDim.x + threadIdx.x;
  if (i < n) out[i] = (_Float16)in[i];
}

__global__ void k_zero_f16(_Float16* p, long n) {
  long i = (long)blockIdx.x * blockDim.x + threadIdx.x;
  if (i < n) p[i] = (_Float16)0.0f;
}

// im2col with fused BN(n1): col[b*PP+p][c*9+dy*3+dx], stride 3, pad 1
__global__ void k_im2col_bn(const float* __restrict__ x, const float* g,
                            const float* b, const float* m, const float* v,
                            _Float16* __restrict__ col) {
  long i = (long)blockIdx.x * blockDim.x + threadIdx.x;
  if (i >= (long)M1 * K1) return;
  int k   = (int)(i % K1);
  long rw = i / K1;
  int b_  = (int)(rw / PP), p = (int)(rw % PP);
  int hc = p / WCC, wc = p % WCC;
  int c = k / 9, rem = k % 9, dy = rem / 3, dx = rem % 3;
  int t = hc * 3 - 1 + dy, j = wc * 3 - 1 + dx;
  float val = 0.0f;
  if ((unsigned)t < (unsigned)TT && (unsigned)j < (unsigned)JJ)
    val = bn_ap(x[(((long)b_ * CC + c) * TT + t) * JJ + j], g, b, m, v, c);
  col[rw * K1 + k] = (_Float16)val;
}

// qkv GEMM: out channel o -> {q,k,vT} with fused BN(qbn)+GELU
__global__ void k_gemm_qkv(const _Float16* __restrict__ A,
                           const _Float16* __restrict__ Bw,
                           const float* g, const float* bb, const float* mm,
                           const float* vv, _Float16* __restrict__ qb,
                           _Float16* __restrict__ kb, _Float16* __restrict__ vt) {
  v8f cacc[2][2]; int m0, n0;
  gemm_block(A, Bw, K1, cacc, m0, n0);
  const int lane = threadIdx.x & 31, hi = (lane >> 4) & 1, ln = lane & 15;
#pragma unroll
  for (int mi = 0; mi < 2; ++mi)
#pragma unroll
    for (int ni = 0; ni < 2; ++ni) {
      v8f& cc = cacc[mi][ni];
      const int o = n0 + ni * 16 + ln;          // 0..767
      const int s = o >> 8, ch = o & 255, hD = ch >> 5, hd = ch & 31;
#pragma unroll
      for (int v = 0; v < 8; ++v) {
        const int mg = m0 + mi * 16 + v + 8 * hi;
        const int b_ = mg / PP, p = mg % PP;
        const float y = gelu_ex(bn_ap(cc[v], g, bb, mm, vv, o));
        const long bh = (long)b_ * NHH + hD;
        if (s == 0)      qb[(bh * PP + p) * HDD + hd] = (_Float16)y;
        else if (s == 1) kb[(bh * PP + p) * HDD + hd] = (_Float16)y;
        else             vt[(bh * HDD + hd) * PPAD + p] = (_Float16)y;
      }
    }
}

// flash attention: 1 wave per (b, h, 16-row q-tile); HD=32 == one WMMA K-step
__global__ void k_attn(const _Float16* __restrict__ qb,
                       const _Float16* __restrict__ kb,
                       const _Float16* __restrict__ vt,
                       _Float16* __restrict__ ob) {
  const int id = blockIdx.x;
  const int bh = id / 17, qt = id % 17;
  const int b_ = bh / NHH, h = bh % NHH;
  const _Float16* qp = qb + (long)bh * PP * HDD;
  const _Float16* kp = kb + (long)bh * PP * HDD;
  const _Float16* vp = vt + (long)bh * HDD * PPAD;
  const int lane = threadIdx.x & 31, hi = (lane >> 4) & 1, ln = lane & 15;
  const float scale = 0.17677669529663688f;  // 32^-0.5

  __shared__ __align__(4) _Float16 Pb[16 * 32];

  int qrow = qt * 16 + ln; if (qrow > PP - 1) qrow = PP - 1;
  FragAB qa = frag_ld(qp, qrow, HDD, lane);

  v8f a0, a1; float mr[8], lr[8];
#pragma unroll
  for (int i = 0; i < 8; ++i) { a0[i] = 0.f; a1[i] = 0.f; mr[i] = -3e38f; lr[i] = 0.f; }

  for (int kt = 0; kt < PPAD / 32; ++kt) {
    const int kc = kt * 32;
    int k0r = kc + ln;       if (k0r > PP - 1) k0r = PP - 1;
    int k1r = kc + 16 + ln;  if (k1r > PP - 1) k1r = PP - 1;
    FragAB kb0 = frag_ld(kp, k0r, HDD, lane);
    FragAB kb1 = frag_ld(kp, k1r, HDD, lane);
    v8f z;
#pragma unroll
    for (int i = 0; i < 8; ++i) z[i] = 0.f;
    v8f s0 = __builtin_amdgcn_wmma_f32_16x16x32_f16(false, qa.v, false, kb0.v,
                                                    (short)0, z, false, false);
    v8f s1 = __builtin_amdgcn_wmma_f32_16x16x32_f16(false, qa.v, false, kb1.v,
                                                    (short)0, z, false, false);
    const bool ok0 = (kc + ln) < PP, ok1 = (kc + 16 + ln) < PP;
    float p0[8], p1[8], rm[8], rs[8];
#pragma unroll
    for (int v = 0; v < 8; ++v) {
      p0[v] = ok0 ? s0[v] * scale : -3e38f;
      p1[v] = ok1 ? s1[v] * scale : -3e38f;
      rm[v] = fmaxf(p0[v], p1[v]);
    }
#pragma unroll
    for (int msk = 8; msk >= 1; msk >>= 1)
#pragma unroll
      for (int v = 0; v < 8; ++v)
        rm[v] = fmaxf(rm[v], __shfl_xor(rm[v], msk, 32));
#pragma unroll
    for (int v = 0; v < 8; ++v) {
      const float mn = fmaxf(mr[v], rm[v]);
      const float f  = __expf(mr[v] - mn);
      p0[v] = __expf(p0[v] - mn);
      p1[v] = __expf(p1[v] - mn);
      rs[v] = p0[v] + p1[v];
      a0[v] *= f; a1[v] *= f;
      lr[v] = lr[v] * f; mr[v] = mn;
    }
#pragma unroll
    for (int msk = 8; msk >= 1; msk >>= 1)
#pragma unroll
      for (int v = 0; v < 8; ++v)
        rs[v] += __shfl_xor(rs[v], msk, 32);
#pragma unroll
    for (int v = 0; v < 8; ++v) {
      lr[v] += rs[v];
      const int mrow = v + 8 * hi;
      Pb[mrow * 32 + ln]      = (_Float16)p0[v];
      Pb[mrow * 32 + 16 + ln] = (_Float16)p1[v];
    }
    __syncthreads();
    FragAB pa  = frag_ld(Pb, ln, 32, lane);
    FragAB vb0 = frag_ld(vp + kc, ln,      PPAD, lane);  // hd 0..15
    FragAB vb1 = frag_ld(vp + kc, 16 + ln, PPAD, lane);  // hd 16..31
    a0 = __builtin_amdgcn_wmma_f32_16x16x32_f16(false, pa.v, false, vb0.v,
                                                (short)0, a0, false, false);
    a1 = __builtin_amdgcn_wmma_f32_16x16x32_f16(false, pa.v, false, vb1.v,
                                                (short)0, a1, false, false);
    __syncthreads();
  }
#pragma unroll
  for (int v = 0; v < 8; ++v) {
    const int q = qt * 16 + v + 8 * hi;
    if (q < PP) {
      const float inv = 1.0f / lr[v];
      const int c0c = h * 32 + ln, c1c = c0c + 16;
      ob[((long)b_ * CC + c0c) * PP + q] = (_Float16)(a0[v] * inv);
      ob[((long)b_ * CC + c1c) * PP + q] = (_Float16)(a1[v] * inv);
    }
  }
}

// bilinear 43x6 -> 128x17 (half-pixel), to GEMM A layout [b*t*j][c]
__global__ void k_resize(const _Float16* __restrict__ ob,
                         _Float16* __restrict__ rsz) {
  long i = (long)blockIdx.x * blockDim.x + threadIdx.x;
  if (i >= (long)M2 * CC) return;
  const int c = (int)(i % CC);
  long r = i / CC;
  const int j = (int)(r % JJ); r /= JJ;
  const int t = (int)(r % TT);
  const int b_ = (int)(r / TT);
  const float ft = (t + 0.5f) * (43.0f / 128.0f) - 0.5f;
  const float fj = (j + 0.5f) * (6.0f / 17.0f) - 0.5f;
  int t0 = (int)floorf(ft), j0 = (int)floorf(fj);
  const float wt = ft - t0, wj = fj - j0;
  int t1 = t0 + 1, j1 = j0 + 1;
  t0 = min(max(t0, 0), HCC - 1); t1 = min(max(t1, 0), HCC - 1);
  j0 = min(max(j0, 0), WCC - 1); j1 = min(max(j1, 0), WCC - 1);
  const _Float16* src = ob + ((long)b_ * CC + c) * PP;
  const float v00 = (float)src[t0 * WCC + j0], v01 = (float)src[t0 * WCC + j1];
  const float v10 = (float)src[t1 * WCC + j0], v11 = (float)src[t1 * WCC + j1];
  const float val = (1.f - wt) * ((1.f - wj) * v00 + wj * v01) +
                    wt * ((1.f - wj) * v10 + wj * v11);
  rsz[((long)b_ * TT * JJ + t * JJ + j) * CC + c] = (_Float16)val;
}

// proj 1x1 + bias + residual -> x2 (f32, written to d_out)
__global__ void k_gemm_proj(const _Float16* __restrict__ A,
                            const _Float16* __restrict__ Bw,
                            const float* __restrict__ pb,
                            const float* __restrict__ x,
                            float* __restrict__ out) {
  v8f cacc[2][2]; int m0, n0;
  gemm_block(A, Bw, CC, cacc, m0, n0);
  const int lane = threadIdx.x & 31, hi = (lane >> 4) & 1, ln = lane & 15;
#pragma unroll
  for (int mi = 0; mi < 2; ++mi)
#pragma unroll
    for (int ni = 0; ni < 2; ++ni) {
      v8f& cc = cacc[mi][ni];
      const int c = n0 + ni * 16 + ln;
#pragma unroll
      for (int v = 0; v < 8; ++v) {
        const int mg = m0 + mi * 16 + v + 8 * hi;
        const int b_ = mg / (TT * JJ), r = mg % (TT * JJ), t = r / JJ, j = r % JJ;
        const long idx = (((long)b_ * CC + c) * TT + t) * JJ + j;
        out[idx] = cc[v] + pb[c] + x[idx];
      }
    }
}

// bn(n2) of x2 -> f16 GEMM A layout [row][c]
__global__ void k_bn2_f16(const float* __restrict__ x2, const float* g,
                          const float* b, const float* m, const float* v,
                          _Float16* __restrict__ xh2) {
  long i = (long)blockIdx.x * blockDim.x + threadIdx.x;
  if (i >= (long)M2 * CC) return;
  const int c = (int)(i % CC);
  long r = i / CC;
  const int j = (int)(r % JJ); long r2 = r / JJ;
  const int t = (int)(r2 % TT), b_ = (int)(r2 / TT);
  const float val = x2[(((long)b_ * CC + c) * TT + t) * JJ + j];
  xh2[r * CC + c] = (_Float16)bn_ap(val, g, b, m, v, c);
}

// fc1 + bn1 + gelu -> f16 hidden chunk
__global__ void k_gemm_fc1(const _Float16* __restrict__ A,
                           const _Float16* __restrict__ Bw,
                           const float* __restrict__ fb, const float* g,
                           const float* bb, const float* mm, const float* vv,
                           _Float16* __restrict__ hmid) {
  v8f cacc[2][2]; int m0, n0;
  gemm_block(A, Bw, CC, cacc, m0, n0);
  const int lane = threadIdx.x & 31, hi = (lane >> 4) & 1, ln = lane & 15;
#pragma unroll
  for (int mi = 0; mi < 2; ++mi)
#pragma unroll
    for (int ni = 0; ni < 2; ++ni) {
      v8f& cc = cacc[mi][ni];
      const int n = n0 + ni * 16 + ln;
#pragma unroll
      for (int v = 0; v < 8; ++v) {
        const int mg = m0 + mi * 16 + v + 8 * hi;
        hmid[(long)mg * HIDN + n] =
            (_Float16)gelu_ex(bn_ap(cc[v] + fb[n], g, bb, mm, vv, n));
      }
    }
}

// fc2 + bn2 epilogue, accumulate into d_out (final residual add)
__global__ void k_gemm_fc2(const _Float16* __restrict__ A,
                           const _Float16* __restrict__ Bw,
                           const float* __restrict__ fb, const float* g,
                           const float* bb, const float* mm, const float* vv,
                           float* __restrict__ out, int mbase) {
  v8f cacc[2][2]; int m0, n0;
  gemm_block(A, Bw, HIDN, cacc, m0, n0);
  const int lane = threadIdx.x & 31, hi = (lane >> 4) & 1, ln = lane & 15;
#pragma unroll
  for (int mi = 0; mi < 2; ++mi)
#pragma unroll
    for (int ni = 0; ni < 2; ++ni) {
      v8f& cc = cacc[mi][ni];
      const int c = n0 + ni * 16 + ln;
#pragma unroll
      for (int v = 0; v < 8; ++v) {
        const int mg = mbase + m0 + mi * 16 + v + 8 * hi;
        const int b_ = mg / (TT * JJ), r = mg % (TT * JJ), t = r / JJ, j = r % JJ;
        const long idx = (((long)b_ * CC + c) * TT + t) * JJ + j;
        out[idx] += bn_ap(cc[v] + fb[c], g, bb, mm, vv, c);
      }
    }
}

// ------------------------------------------------------------------ launch ---
extern "C" void kernel_launch(void* const* d_in, const int* in_sizes, int n_in,
                              void* d_out, int out_size, void* d_ws,
                              size_t ws_size, hipStream_t stream) {
  const float* x      = (const float*)d_in[0];
  const float* n1g = (const float*)d_in[1],  *n1b = (const float*)d_in[2];
  const float* n1m = (const float*)d_in[3],  *n1v = (const float*)d_in[4];
  const float* qkvw = (const float*)d_in[5];
  const float* qg = (const float*)d_in[6],   *qb_ = (const float*)d_in[7];
  const float* qm = (const float*)d_in[8],   *qv = (const float*)d_in[9];
  const float* projw = (const float*)d_in[10], *projb = (const float*)d_in[11];
  const float* n2g = (const float*)d_in[12], *n2b = (const float*)d_in[13];
  const float* n2m = (const float*)d_in[14], *n2v = (const float*)d_in[15];
  const float* fc1w = (const float*)d_in[16], *fc1b = (const float*)d_in[17];
  const float* b1g = (const float*)d_in[18], *b1b = (const float*)d_in[19];
  const float* b1m = (const float*)d_in[20], *b1v = (const float*)d_in[21];
  const float* fc2w = (const float*)d_in[22], *fc2b = (const float*)d_in[23];
  const float* b2g = (const float*)d_in[24], *b2b = (const float*)d_in[25];
  const float* b2m = (const float*)d_in[26], *b2v = (const float*)d_in[27];
  float* out = (float*)d_out;

  // workspace carve-out (halves), 256B aligned
  char* ws = (char*)d_ws;
  size_t off = 0;
  auto alloc_h = [&](long n) {
    _Float16* p = (_Float16*)(ws + off);
    off = (off + (size_t)n * 2 + 255) & ~(size_t)255;
    return p;
  };
  _Float16* wq   = alloc_h((long)N1 * K1);
  _Float16* wpj  = alloc_h((long)CC * CC);
  _Float16* wf1  = alloc_h((long)HIDN * CC);
  _Float16* wf2  = alloc_h((long)CC * HIDN);
  _Float16* col  = alloc_h((long)M1 * K1);
  _Float16* qbuf = alloc_h((long)BB * NHH * PP * HDD);
  _Float16* kbuf = alloc_h((long)BB * NHH * PP * HDD);
  _Float16* vt   = alloc_h((long)BB * NHH * HDD * PPAD);
  _Float16* ob   = alloc_h((long)BB * CC * PP);
  _Float16* rsz  = alloc_h((long)M2 * CC);
  _Float16* xh2  = alloc_h((long)M2 * CC);
  _Float16* hmid = alloc_h((long)MCH * HIDN);

  auto nb = [](long n) { return (unsigned)((n + 255) / 256); };

  // weight casts f32 -> f16 (layouts already match [N][K])
  k_cast_f16<<<nb((long)N1 * K1), 256, 0, stream>>>(qkvw, wq, (long)N1 * K1);
  k_cast_f16<<<nb((long)CC * CC), 256, 0, stream>>>(projw, wpj, (long)CC * CC);
  k_cast_f16<<<nb((long)HIDN * CC), 256, 0, stream>>>(fc1w, wf1, (long)HIDN * CC);
  k_cast_f16<<<nb((long)CC * HIDN), 256, 0, stream>>>(fc2w, wf2, (long)CC * HIDN);
  k_zero_f16<<<nb((long)BB * NHH * HDD * PPAD), 256, 0, stream>>>(
      vt, (long)BB * NHH * HDD * PPAD);

  // stage 1: BN + im2col, qkv GEMM (BN+GELU fused), attention
  k_im2col_bn<<<nb((long)M1 * K1), 256, 0, stream>>>(x, n1g, n1b, n1m, n1v, col);
  k_gemm_qkv<<<dim3(M1 / 64, N1 / 128), 256, 0, stream>>>(
      col, wq, qg, qb_, qm, qv, qbuf, kbuf, vt);
  k_attn<<<BB * NHH * 17, 32, 0, stream>>>(qbuf, kbuf, vt, ob);

  // stage 2: bilinear upsample, proj + residual -> x2 in d_out
  k_resize<<<nb((long)M2 * CC), 256, 0, stream>>>(ob, rsz);
  k_gemm_proj<<<dim3(M2 / 64, CC / 128), 256, 0, stream>>>(rsz, wpj, projb, x, out);

  // stage 3: MLP (chunked over M to bound the hidden-activation scratch)
  k_bn2_f16<<<nb((long)M2 * CC), 256, 0, stream>>>(out, n2g, n2b, n2m, n2v, xh2);
  for (int ch = 0; ch < NCHK; ++ch) {
    const int base = ch * MCH;
    k_gemm_fc1<<<dim3(MCH / 64, HIDN / 128), 256, 0, stream>>>(
        xh2 + (long)base * CC, wf1, fc1b, b1g, b1b, b1m, b1v, hmid);
    k_gemm_fc2<<<dim3(MCH / 64, CC / 128), 256, 0, stream>>>(
        hmid, wf2, fc2b, b2g, b2b, b2m, b2v, out, base);
  }
  (void)in_sizes; (void)n_in; (void)out_size; (void)ws_size;
}
